// MHA_76845554860415
// MI455X (gfx1250) — matrix operand order repaired and checked by reading, not compiled, not run on hardware
//
#include <hip/hip_runtime.h>
#include <hip/hip_bf16.h>
#include <stdint.h>

// ---------------------------------------------------------------------------
// MHA forward for B=2,S=2048,C=1024,H=8,DK=64,DV=192 on gfx1250 (MI455X).
// All GEMM-like math via v_wmma_f32_16x16x32_bf16; attention fully fused
// (flash-style) so the S x S score matrix never hits HBM.
// v4: projection GEMMs now stage the shared W tile in LDS via double-buffered
//     global_load_async_to_lds_b128 (ASYNCcnt-pipelined), removing the 4x
//     redundant W streaming; flash kernel unchanged from v3.
// ---------------------------------------------------------------------------

typedef __bf16 bf16_t;
typedef __bf16 bf16x8  __attribute__((ext_vector_type(8)));
typedef __bf16 bf16x16 __attribute__((ext_vector_type(16)));
typedef float  f32x8   __attribute__((ext_vector_type(8)));

#define WMMA_BF16(a, b, c) \
  __builtin_amdgcn_wmma_f32_16x16x32_bf16(false, (a), false, (b), (short)0, (c), false, false)

// ---------------------------- constants ------------------------------------
constexpr int kB  = 2;
constexpr int kS  = 2048;
constexpr int kC  = 1024;
constexpr int kH  = 8;
constexpr int kDK = 64;
constexpr int kDV = 192;
constexpr int kHD = kH * kDK;   // 512
constexpr int kHV = kH * kDV;   // 1536
constexpr int kM  = kB * kS;    // 4096 rows for all projections

// ---------------- 16-lane xor butterfly via v_permlane16_b32 ----------------
__device__ __forceinline__ float xor_lane16(float v, int selLo, int selHi) {
  int i = __float_as_int(v);
  return __int_as_float(__builtin_amdgcn_permlane16(i, i, selLo, selHi, false, false));
}
// selector constants: nibble i = i ^ mask
#define SEL1_LO  0x67452301
#define SEL1_HI  0xEFCDAB89
#define SEL2_LO  0x54761032
#define SEL2_HI  0xDCFE98BA
#define SEL4_LO  0x32107654
#define SEL4_HI  0xBA98FEDC
#define SEL8_LO  0xFEDCBA98
#define SEL8_HI  0x76543210

__device__ __forceinline__ float reduce16_max(float v) {
  v = fmaxf(v, xor_lane16(v, SEL1_LO, SEL1_HI));
  v = fmaxf(v, xor_lane16(v, SEL2_LO, SEL2_HI));
  v = fmaxf(v, xor_lane16(v, SEL4_LO, SEL4_HI));
  v = fmaxf(v, xor_lane16(v, SEL8_LO, SEL8_HI));
  return v;
}
__device__ __forceinline__ float reduce16_sum(float v) {
  v += xor_lane16(v, SEL1_LO, SEL1_HI);
  v += xor_lane16(v, SEL2_LO, SEL2_HI);
  v += xor_lane16(v, SEL4_LO, SEL4_HI);
  v += xor_lane16(v, SEL8_LO, SEL8_HI);
  return v;
}

// ---------------- async 16B global -> LDS copy (CDNA5) ----------------------
__device__ __forceinline__ void async_b128(uint32_t lds_addr, const bf16_t* g) {
  asm volatile("global_load_async_to_lds_b128 %0, %1, off"
               :: "v"(lds_addr), "v"(g) : "memory");
}

// ------------------------- fp32 -> bf16 convert -----------------------------
__global__ void mha_cvt_bf16(const float* __restrict__ in, bf16_t* __restrict__ out, int n) {
  int i = blockIdx.x * blockDim.x + threadIdx.x;
  if (i < n) out[i] = (bf16_t)in[i];
}

// --------------------------- generic WMMA GEMM ------------------------------
// Y[M][N] = A[M][K] (bf16, row-major) @ W[N][K]^T (bf16, row-major) + bias[N]
// OUT_MODE 0: f32 [M][N]; 1: bf16 [M][N]; 2: bf16 transposed per batch.
// Block 128 = 4 waves; WG tile 64(M) x 64(N). The W tile (shared by all 4
// waves) is double-buffered in LDS via async global->LDS loads.
// Requires: M % 64 == 0, N % 64 == 0, K % 32 == 0.
template <int OUT_MODE>
__global__ __launch_bounds__(128, 1)
void mha_gemm_wmma(const bf16_t* __restrict__ A, const bf16_t* __restrict__ W,
                   const float* __restrict__ bias, void* __restrict__ Y,
                   int M, int N, int K, int Sdim, int Ntot) {
  const int tid  = threadIdx.x;
  const int lane = tid & 31;
  const int wave = tid >> 5;                  // 0..3
  const int nl   = lane & 15;
  const int kh   = lane >> 4;
  const int m0   = (blockIdx.y * 4 + wave) * 16;
  const int n0   = blockIdx.x * 64;

  __shared__ __align__(32) bf16_t wtile[2][64][32];   // 2 x 4 KB: [n][k-chunk]

  // cooperative async fill: 64 rows x 32 elems = 256 x 16B chunks, 2/thread
  const bf16_t* wSrc0[2];
  uint32_t wDst0[2];
#pragma unroll
  for (int j = 0; j < 2; ++j) {
    const int c = tid + 128 * j, row = c >> 2, cg = c & 3;
    wSrc0[j] = W + (size_t)(n0 + row) * K + cg * 8;
    wDst0[j] = (uint32_t)(size_t)&wtile[0][row][cg * 8];
  }
  auto issue_tile = [&](int tile) {
    const uint32_t bs = (tile & 1) * (uint32_t)sizeof(wtile[0]);
#pragma unroll
    for (int j = 0; j < 2; ++j)
      async_b128(wDst0[j] + bs, wSrc0[j] + tile * 32);
  };

  const bf16_t* aRow = A + (size_t)(m0 + nl) * K;
  const int NT = K / 32;

  f32x8 acc[4] = {};
  issue_tile(0);
  for (int it = 0; it < NT; ++it) {
    if (it + 1 < NT) {
      issue_tile(it + 1);
      asm volatile("s_wait_asynccnt 0x2" ::: "memory");  // tile `it` landed
    } else {
      asm volatile("s_wait_asynccnt 0x0" ::: "memory");
    }
    __syncthreads();

    const int k0 = it * 32;
    bf16x16 aF;
    {
      union { bf16x16 v; bf16x8 h[2]; } u;
      u.h[0] = *(const bf16x8*)(aRow + k0 + 16 * kh);
      u.h[1] = *(const bf16x8*)(aRow + k0 + 16 * kh + 8);
      aF = u.v;
    }
    const bf16_t(*wt_)[32] = wtile[it & 1];
#pragma unroll
    for (int t = 0; t < 4; ++t) {
      bf16x16 bF = *(const bf16x16*)&wt_[16 * t + nl][16 * kh];
      acc[t] = WMMA_BF16(aF, bF, acc[t]);
    }
    __syncthreads();    // all waves done with buffer (it&1) before refill
  }

#pragma unroll
  for (int t = 0; t < 4; ++t) {
    const int n  = n0 + 16 * t + nl;
    const float bv = bias[n];
#pragma unroll
    for (int r = 0; r < 8; ++r) {
      const int m = m0 + r + 8 * kh;
      const float val = acc[t][r] + bv;
      if (OUT_MODE == 0) {
        ((float*)Y)[(size_t)m * N + n] = val;
      } else if (OUT_MODE == 1) {
        ((bf16_t*)Y)[(size_t)m * N + n] = (bf16_t)val;
      } else {
        const int b = m / Sdim, s = m % Sdim;
        ((bf16_t*)Y)[((size_t)b * Ntot + n) * Sdim + s] = (bf16_t)val;
      }
    }
  }
}

// ------------------------- fused flash attention ----------------------------
// Q,K bf16 [B*S][H*DK]; Vt bf16 [B][H*DV][S]; Sc bf16 [B*S][H*DV]
// Grid (S/64, H, B); block 256 (8 waves). Waves pair on a 16-row q-tile;
// each wave owns 96 DV cols. K/V tiles double-buffered in LDS via async
// global->LDS loads (shared by all 8 waves).
__global__ __launch_bounds__(256, 1)
void mha_flash_attn_wmma(const bf16_t* __restrict__ Qb,
                         const bf16_t* __restrict__ Kb,
                         const bf16_t* __restrict__ Vt,
                         bf16_t* __restrict__ Sc) {
  const float scale = 0.125f;                 // 1/sqrt(DK)
  const int tid   = threadIdx.x;
  const int lane  = tid & 31;
  const int wave  = tid >> 5;                 // 0..7
  const int qtile = wave >> 1;                // 0..3
  const int dvh   = wave & 1;                 // 0..1
  const int nl    = lane & 15;
  const int kh    = lane >> 4;
  const int b     = blockIdx.z;
  const int h     = blockIdx.y;
  const int q0    = blockIdx.x * 64 + qtile * 16;

  __shared__ __align__(32) bf16_t kbuf[2][32][64];    //  8 KB: [key][dk]
  __shared__ __align__(32) bf16_t vbuf[2][192][32];   // 24 KB: [dv][key]
  __shared__ __align__(32) bf16_t pbuf[8][16][32];    //  8 KB: per-wave P

  // ---- cooperative async copy setup (256 threads) ----
  const int krow = tid >> 2, kcg = tid & 3;
  const bf16_t* kSrc0 = Kb + (size_t)(b * kS + krow) * kHD + h * kDK + kcg * 8;
  const uint32_t kDst0 = (uint32_t)(size_t)&kbuf[0][krow][kcg * 8];
  const bf16_t* vSrc0[3];
  uint32_t vDst0[3];
#pragma unroll
  for (int j = 0; j < 3; ++j) {
    const int c = tid + 256 * j, vrow = c >> 2, vcg = c & 3;
    vSrc0[j] = Vt + ((size_t)b * kHV + h * kDV + vrow) * kS + vcg * 8;
    vDst0[j] = (uint32_t)(size_t)&vbuf[0][vrow][vcg * 8];
  }
  auto issue_tile = [&](int tile) {
    const int bs = tile & 1;
    async_b128(kDst0 + bs * (uint32_t)sizeof(kbuf[0]), kSrc0 + (size_t)tile * 32 * kHD);
#pragma unroll
    for (int j = 0; j < 3; ++j)
      async_b128(vDst0[j] + bs * (uint32_t)sizeof(vbuf[0]), vSrc0[j] + tile * 32);
  };

  // ---- resident Q fragments (two 32-wide chunks of DK=64) ----
  const bf16_t* qRow = Qb + (size_t)(b * kS + q0 + nl) * kHD + h * kDK;
  bf16x16 qF[2];
#pragma unroll
  for (int kk = 0; kk < 2; ++kk) {
    union { bf16x16 v; bf16x8 hh[2]; } u;
    u.hh[0] = *(const bf16x8*)(qRow + kk * 32 + 16 * kh);
    u.hh[1] = *(const bf16x8*)(qRow + kk * 32 + 16 * kh + 8);
    qF[kk] = u.v;
  }

  float mrow[8], lrow[8];
#pragma unroll
  for (int r = 0; r < 8; ++r) { mrow[r] = -1e30f; lrow[r] = 0.0f; }
  f32x8 oAcc[6] = {};                          // 16 x 96 f32 accumulator

  constexpr int NT = kS / 32;                  // 64 k-tiles
  issue_tile(0);

  for (int it = 0; it < NT; ++it) {
    if (it + 1 < NT) {
      issue_tile(it + 1);
      asm volatile("s_wait_asynccnt 0x4" ::: "memory");  // tile `it` landed
    } else {
      asm volatile("s_wait_asynccnt 0x0" ::: "memory");
    }
    __syncthreads();                            // tile visible to all waves

    const bf16_t(*kt_)[64] = kbuf[it & 1];
    const bf16_t(*vt_)[32] = vbuf[it & 1];

    // ---- S = Q K^T for a 16 x 32 tile (4 WMMAs, operands from LDS) ----
    f32x8 sAcc[2] = {};
#pragma unroll
    for (int t = 0; t < 2; ++t) {
#pragma unroll
      for (int kk = 0; kk < 2; ++kk) {
        bf16x16 bF = *(const bf16x16*)&kt_[16 * t + nl][kk * 32 + 16 * kh];
        sAcc[t] = WMMA_BF16(qF[kk], bF, sAcc[t]);
      }
    }

    // ---- online softmax update (row = r + 8*kh, cols nl and 16+nl) ----
#pragma unroll
    for (int r = 0; r < 8; ++r) {
      float s0 = sAcc[0][r] * scale;
      float s1 = sAcc[1][r] * scale;
      float mx = reduce16_max(fmaxf(s0, s1));
      const float mnew  = fmaxf(mrow[r], mx);
      const float alpha = __expf(mrow[r] - mnew);
      const float p0 = __expf(s0 - mnew);
      const float p1 = __expf(s1 - mnew);
      const float rs = reduce16_sum(p0 + p1);
      lrow[r] = lrow[r] * alpha + rs;
      mrow[r] = mnew;
#pragma unroll
      for (int t = 0; t < 6; ++t) oAcc[t][r] *= alpha;
      pbuf[wave][r + 8 * kh][nl]      = (bf16_t)p0;
      pbuf[wave][r + 8 * kh][16 + nl] = (bf16_t)p1;
    }

    // ---- re-read P in A-fragment layout (same-wave DS ops in-order) ----
    bf16x16 pF;
    {
      union { bf16x16 v; bf16x8 hh[2]; } u;
      u.hh[0] = *(const bf16x8*)(&pbuf[wave][nl][16 * kh]);
      u.hh[1] = *(const bf16x8*)(&pbuf[wave][nl][16 * kh + 8]);
      pF = u.v;
    }

    // ---- O += P V (6 WMMAs, V operand from LDS, this wave's DV half) ----
#pragma unroll
    for (int t = 0; t < 6; ++t) {
      bf16x16 vF = *(const bf16x16*)&vt_[dvh * 96 + 16 * t + nl][16 * kh];
      oAcc[t] = WMMA_BF16(pF, vF, oAcc[t]);
    }

    __syncthreads();   // everyone done with buffer (it&1) before it is refilled
  }

  // ---- normalize and emit bf16 scores [B*S][H*DV] ----
#pragma unroll
  for (int r = 0; r < 8; ++r) {
    const float inv = lrow[r] > 0.0f ? 1.0f / lrow[r] : 0.0f;
    const int m = q0 + r + 8 * kh;
    bf16_t* outRow = Sc + (size_t)(b * kS + m) * kHV + h * kDV + dvh * 96;
#pragma unroll
    for (int t = 0; t < 6; ++t)
      outRow[16 * t + nl] = (bf16_t)(oAcc[t][r] * inv);
  }
}

// ------------------------------- launcher -----------------------------------
extern "C" void kernel_launch(void* const* d_in, const int* in_sizes, int n_in,
                              void* d_out, int out_size, void* d_ws, size_t ws_size,
                              hipStream_t stream) {
  (void)in_sizes; (void)n_in; (void)out_size; (void)ws_size;

  const float* x  = (const float*)d_in[0];
  const float* Wq = (const float*)d_in[1];
  const float* bq = (const float*)d_in[2];
  const float* Wk = (const float*)d_in[3];
  const float* bk = (const float*)d_in[4];
  const float* Wv = (const float*)d_in[5];
  const float* bv = (const float*)d_in[6];
  const float* Wo = (const float*)d_in[7];
  const float* bo = (const float*)d_in[8];
  float* out = (float*)d_out;

  // Workspace carve-up (bytes; every size is already 256-aligned). ~48 MB.
  char* ws = (char*)d_ws;
  size_t off = 0;
  auto carve = [&](size_t bytes) { char* p = ws + off; off += bytes; return p; };
  bf16_t* xb  = (bf16_t*)carve((size_t)kM * kC * 2);       //  8 MB
  bf16_t* wqb = (bf16_t*)carve((size_t)kHD * kC * 2);      //  1 MB
  bf16_t* wkb = (bf16_t*)carve((size_t)kHD * kC * 2);      //  1 MB
  bf16_t* wvb = (bf16_t*)carve((size_t)kHV * kC * 2);      //  3 MB
  bf16_t* wob = (bf16_t*)carve((size_t)kC * kHV * 2);      //  3 MB
  bf16_t* Qb  = (bf16_t*)carve((size_t)kM * kHD * 2);      //  4 MB
  bf16_t* Kb  = (bf16_t*)carve((size_t)kM * kHD * 2);      //  4 MB
  bf16_t* Vt  = (bf16_t*)carve((size_t)kB * kHV * kS * 2); // 12 MB
  bf16_t* Scb = (bf16_t*)carve((size_t)kM * kHV * 2);      // 12 MB

  auto cvt = [&](const float* src, bf16_t* dst, int n) {
    mha_cvt_bf16<<<dim3((n + 255) / 256), dim3(256), 0, stream>>>(src, dst, n);
  };
  cvt(x,  xb,  kM * kC);
  cvt(Wq, wqb, kHD * kC);
  cvt(Wk, wkb, kHD * kC);
  cvt(Wv, wvb, kHV * kC);
  cvt(Wo, wob, kC * kHV);

  const dim3 blk(128);
  // Q = x Wq^T + bq  (bf16 out)
  mha_gemm_wmma<1><<<dim3(kHD / 64, kM / 64), blk, 0, stream>>>(
      xb, wqb, bq, Qb, kM, kHD, kC, kS, kHD);
  // K = x Wk^T + bk  (bf16 out)
  mha_gemm_wmma<1><<<dim3(kHD / 64, kM / 64), blk, 0, stream>>>(
      xb, wkb, bk, Kb, kM, kHD, kC, kS, kHD);
  // V = x Wv^T + bv, stored transposed per batch: Vt[b][n][s]
  mha_gemm_wmma<2><<<dim3(kHV / 64, kM / 64), blk, 0, stream>>>(
      xb, wvb, bv, Vt, kM, kHV, kC, kS, kHV);

  // Fused flash attention -> bf16 scores [B*S][H*DV]
  mha_flash_attn_wmma<<<dim3(kS / 64, kH, kB), dim3(256), 0, stream>>>(Qb, Kb, Vt, Scb);

  // out = scores Wo^T + bo  (f32 out)
  mha_gemm_wmma<0><<<dim3(kC / 64, kM / 64), blk, 0, stream>>>(
      Scb, wob, bo, out, kM, kC, kHV, kS, kC);
}
